// TreeDecoderAttn_71914932404603
// MI455X (gfx1250) — matrix operand order repaired and checked by reading, not compiled, not run on hardware
//
#include <hip/hip_runtime.h>

// ---------------------------------------------------------------------------
// Types for CDNA5 WMMA
// ---------------------------------------------------------------------------
typedef __bf16 bfloat;
typedef __attribute__((ext_vector_type(8)))  __bf16 v8bf;
typedef __attribute__((ext_vector_type(16))) __bf16 v16bf;
typedef __attribute__((ext_vector_type(8)))  float  v8f;

#define NEG_INF (-__builtin_inff())

// Problem constants
#define BB  16
#define XLEN 256
#define YLEN 128
#define DD  512
#define DVV 512
#define WVN 32000
#define RVN 200
#define VTOT 32200
#define INV_SCALE (1.0f/22.627416997969522f)   // 1/sqrt(512)

// ---------------------------------------------------------------------------
// gfx1250 async global->LDS copy (ASYNCcnt path), guarded so the build never
// breaks if the toolchain lacks the builtin.
// Builtin signature (from hipcc diagnostic): (int4 AS1*, int4 AS3*, Imm, Imm).
// ---------------------------------------------------------------------------
#if defined(__HIP_DEVICE_COMPILE__) && defined(__gfx1250__) && \
    __has_builtin(__builtin_amdgcn_global_load_async_to_lds_b128)
#define HAS_ASYNC_LDS 1
#else
#define HAS_ASYNC_LDS 0
#endif

#if HAS_ASYNC_LDS
typedef int v4i_ __attribute__((vector_size(16)));
typedef __attribute__((address_space(1))) v4i_* g_v4i;
typedef __attribute__((address_space(3))) v4i_* l_v4i;
__device__ __forceinline__ void async_copy_b128(const void* gsrc, void* ldst) {
  // AS1 ptr = 64-bit raw address; AS3 ptr = 32-bit LDS offset (low 32 bits of
  // the generic LDS address per the aperture mapping).
  __builtin_amdgcn_global_load_async_to_lds_b128(
      (g_v4i)(unsigned long long)gsrc,
      (l_v4i)(unsigned int)(unsigned long long)ldst, 0, 0);
}
__device__ __forceinline__ void wait_async0() {
#if __has_builtin(__builtin_amdgcn_s_wait_asynccnt)
  __builtin_amdgcn_s_wait_asynccnt(0);
#else
  asm volatile("s_wait_asynccnt 0x0" ::: "memory");
#endif
}
#endif

__device__ __forceinline__ v8f wmma_bf16(v16bf a, v16bf b, v8f c) {
  return __builtin_amdgcn_wmma_f32_16x16x32_bf16(false, a, false, b, (short)0, c,
                                                 false, false);
}

// A fragment 16x32 bf16 (M=16 rows = batch, K=32): lanes 0-15 hold row M=lane,
// K k0..k0+15; lanes 16-31 hold row M=lane-16, K k0+16..k0+31.
__device__ __forceinline__ v16bf load_frag_a(const bfloat* __restrict__ base,
                                             int stride, int k0, int lane) {
  int row = lane & 15;
  int sel = (lane >> 4) & 1;
  const bfloat* p = base + (size_t)row * stride + k0 + sel * 16;
  v8bf lo = *(const v8bf*)(p);
  v8bf hi = *(const v8bf*)(p + 8);
  return __builtin_shufflevector(lo, hi, 0,1,2,3,4,5,6,7,8,9,10,11,12,13,14,15);
}

// B fragment 32x16 bf16 where B[k][n] = W[n][k] (row-major W, contiguous in k):
// lanes 0-15: column n = nbase+lane, K k0..k0+15 (one 32B load);
// lanes 16-31: column n = nbase+lane-16, K k0+16..k0+31.
__device__ __forceinline__ v16bf load_frag_b(const bfloat* __restrict__ W,
                                             int stride, int k0, int lane, int nbase) {
  int n = nbase + (lane & 15);
  int sel = (lane >> 4) & 1;
  const bfloat* p = W + (size_t)n * stride + k0 + sel * 16;
  return *(const v16bf*)p;
}

__device__ __forceinline__ float sigmoidf_(float x) {
  return 1.0f / (1.0f + __expf(-x));
}

// ---------------------------------------------------------------------------
// Utility kernels
// ---------------------------------------------------------------------------
__global__ void cvt_bf16_kernel(bfloat* __restrict__ dst,
                                const float* __restrict__ src, long n) {
  for (long i = (long)blockIdx.x * blockDim.x + threadIdx.x; i < n;
       i += (long)gridDim.x * blockDim.x)
    dst[i] = (bfloat)src[i];
}

__global__ void zero_f32_kernel(float* __restrict__ dst, long n) {
  for (long i = (long)blockIdx.x * blockDim.x + threadIdx.x; i < n;
       i += (long)gridDim.x * blockDim.x)
    dst[i] = 0.0f;
}

__global__ void smask_init_kernel(float* __restrict__ rsm, float* __restrict__ wsm) {
  int i = blockIdx.x * blockDim.x + threadIdx.x;  // 2048 entries
  if (i < BB * YLEN) {
    int s = i & (YLEN - 1);
    float v = (s == 0) ? 0.0f : 1.0f;
    rsm[i] = v;
    wsm[i] = v;
  }
}

// ---------------------------------------------------------------------------
// Per-step input staging: build bf16 LSTM inputs, bf16 h, update smasks
// ---------------------------------------------------------------------------
__global__ void prep_kernel(const float* __restrict__ emb_W,
                            const int* __restrict__ y_train,
                            const float* __restrict__ rule_feed,
                            const float* __restrict__ word_feed,
                            const float* __restrict__ r2w_feed,
                            const float* __restrict__ w2r_feed,
                            const float* __restrict__ rh,
                            const float* __restrict__ wh,
                            bfloat* __restrict__ word_in,
                            bfloat* __restrict__ rule_in,
                            bfloat* __restrict__ wh_bf,
                            bfloat* __restrict__ rh_bf,
                            float* __restrict__ r_smask,
                            float* __restrict__ w_smask, int t) {
  int i = blockIdx.x * blockDim.x + threadIdx.x;
  if (i < 32768) {                       // word_in = [emb, word_feed, w2r_feed]
    int b = i >> 11, c = i & 2047;
    float v;
    if (c < 512) {
      int tok = y_train[b * (YLEN * 3) + t * 3];
      tok = (tok < 0) ? 0 : (tok >= VTOT ? VTOT - 1 : tok);
      v = emb_W[(size_t)tok * DVV + c];
    } else if (c < 1536) v = word_feed[b * 1024 + (c - 512)];
    else                 v = w2r_feed[b * 512 + (c - 1536)];
    word_in[i] = (bfloat)v;
  } else if (i < 65536) {                // rule_in = [emb, rule_feed, r2w_feed]
    int j = i - 32768;
    int b = j >> 11, c = j & 2047;
    float v;
    if (c < 512) {
      int tok = y_train[b * (YLEN * 3) + t * 3];
      tok = (tok < 0) ? 0 : (tok >= VTOT ? VTOT - 1 : tok);
      v = emb_W[(size_t)tok * DVV + c];
    } else if (c < 1536) v = rule_feed[b * 1024 + (c - 512)];
    else                 v = r2w_feed[b * 512 + (c - 1536)];
    rule_in[j] = (bfloat)v;
  } else if (i < 73728) {
    int j = i - 65536; wh_bf[j] = (bfloat)wh[j];
  } else if (i < 81920) {
    int j = i - 73728; rh_bf[j] = (bfloat)rh[j];
  } else if (i < 81952) {
    int j = i - 81920;
    int b = j & 15;
    if (j < 16) {
      if (t > 0) {
        float wm = (float)y_train[b * (YLEN * 3) + t * 3 + 2];
        w_smask[b * YLEN + t] = 1.0f - wm;
      }
    } else {
      r_smask[b * YLEN + t] = 0.0f;
    }
  }
}

// ---------------------------------------------------------------------------
// LSTM step: WMMA gates (M=16 batch x N=16 cols, K=2048 input + 512 hidden),
// then fused sigmoid/tanh cell update + word-mask blend + state scatter.
// Grid: 32 blocks (column tiles of D). Block: 128 threads = 4 waves, wave g
// computes gate g (i,f,g,o).  The 16x2048 bf16 A-panel is staged once per
// block into LDS via async global->LDS copies (shared by all 4 waves).
// ---------------------------------------------------------------------------
__global__ __launch_bounds__(128) void lstm_kernel(
    const bfloat* __restrict__ in_bf,   // 16 x 2048
    const bfloat* __restrict__ h_bf,    // 16 x 512
    const bfloat* __restrict__ wih,     // 2048 x 2048 (N-major rows)
    const bfloat* __restrict__ whh,     // 2048 x 512
    const float* __restrict__ bih, const float* __restrict__ bhh,
    float* __restrict__ h_state, float* __restrict__ c_state,  // 16 x 512
    float* __restrict__ states,         // 16 x 128 x 512
    bfloat* __restrict__ inp_bf, int inp_off,
    const int* __restrict__ y_train, int t, int is_word) {
  __shared__ float sg[4][16][16];
  const int wv = threadIdx.x >> 5;
  const int lane = threadIdx.x & 31;
  const int jt = blockIdx.x;             // column tile within D
  const int nbase = wv * DD + jt * 16;   // gate wv columns

#if HAS_ASYNC_LDS
  __shared__ bfloat sIn[16 * 2048];      // 64 KB A-panel
  {
    const char* g = (const char*)in_bf;
    char* l = (char*)sIn;
#pragma unroll
    for (int it = 0; it < 32; ++it) {    // 128 thr x 32 x 16B = 64 KB
      int off = (threadIdx.x + it * 128) * 16;
      async_copy_b128(g + off, l + off);
    }
    wait_async0();
  }
  __syncthreads();
  const bfloat* Ain = sIn;
#else
  const bfloat* Ain = in_bf;
#endif

  v8f acc = {};
  for (int k0 = 0; k0 < 2048; k0 += 32) {
    v16bf a = load_frag_a(Ain, 2048, k0, lane);
    v16bf b = load_frag_b(wih, 2048, k0, lane, nbase);
    acc = wmma_bf16(a, b, acc);
  }
  for (int k0 = 0; k0 < 512; k0 += 32) {
    v16bf a = load_frag_a(h_bf, 512, k0, lane);
    v16bf b = load_frag_b(whh, 512, k0, lane, nbase);
    acc = wmma_bf16(a, b, acc);
  }
  const int sel = lane >> 4, nc = lane & 15;
#pragma unroll
  for (int r = 0; r < 8; ++r) sg[wv][r + 8 * sel][nc] = acc[r];
  __syncthreads();

  for (int e = threadIdx.x; e < 256; e += 128) {
    int b = e >> 4, jc = e & 15;
    int col = jt * 16 + jc;
    float gi = sg[0][b][jc] + bih[col]            + bhh[col];
    float gf = sg[1][b][jc] + bih[DD + col]       + bhh[DD + col];
    float gg = sg[2][b][jc] + bih[2 * DD + col]   + bhh[2 * DD + col];
    float go = sg[3][b][jc] + bih[3 * DD + col]   + bhh[3 * DD + col];
    float c_old = c_state[b * DD + col];
    float h_old = h_state[b * DD + col];
    float cn = sigmoidf_(gf) * c_old + sigmoidf_(gi) * tanhf(gg);
    float hn = sigmoidf_(go) * tanhf(cn);
    if (is_word) {
      float wm = (float)y_train[b * (YLEN * 3) + t * 3 + 2];
      hn = hn * wm + h_old * (1.0f - wm);
      cn = cn * wm + c_old * (1.0f - wm);
    }
    c_state[b * DD + col] = cn;
    h_state[b * DD + col] = hn;
    states[(size_t)b * YLEN * DD + (size_t)t * DD + col] = hn;
    inp_bf[b * 3072 + inp_off + col] = (bfloat)hn;
  }
}

// ---------------------------------------------------------------------------
// 4 attentions per step: kind 0 rule->enc, 1 word->enc, 2 r2w(self), 3 w2r(self)
// grid 64 = 4 kinds x 16 batches; block 256.
// ---------------------------------------------------------------------------
__global__ __launch_bounds__(256) void attn_kernel(
    const float* __restrict__ rh, const float* __restrict__ wh,
    const float* __restrict__ x_enc, const float* __restrict__ x_enc_k,
    const float* __restrict__ x_mask,
    const float* __restrict__ r_states, const float* __restrict__ w_states,
    const float* __restrict__ r_smask, const float* __restrict__ w_smask,
    float* __restrict__ rule_feed, float* __restrict__ word_feed,
    float* __restrict__ r2w_feed, float* __restrict__ w2r_feed,
    bfloat* __restrict__ inp_bf) {
  __shared__ float sQ[512];
  __shared__ float sW[256];
  __shared__ float sRed[256];
  const int kind = blockIdx.x >> 4;
  const int b = blockIdx.x & 15;
  const int tid = threadIdx.x;

  const float* q = (kind == 0 || kind == 2) ? (rh + b * DD) : (wh + b * DD);
  int S;
  const float* K;
  const float* Vv;
  const float* mask;
  int vstride, Dout;
  if (kind < 2) {
    S = XLEN;
    K = x_enc_k + (size_t)b * XLEN * DD;
    Vv = x_enc + (size_t)b * XLEN * (2 * DD);
    mask = x_mask + b * XLEN;
    vstride = 2 * DD; Dout = 2 * DD;
  } else if (kind == 2) {
    S = YLEN;
    K = w_states + (size_t)b * YLEN * DD;
    Vv = K; mask = w_smask + b * YLEN; vstride = DD; Dout = DD;
  } else {
    S = YLEN;
    K = r_states + (size_t)b * YLEN * DD;
    Vv = K; mask = r_smask + b * YLEN; vstride = DD; Dout = DD;
  }

  for (int d = tid; d < DD; d += 256) sQ[d] = q[d];
  __syncthreads();

  float sc = NEG_INF;
  if (tid < S) {
    const float* kr = K + (size_t)tid * DD;
    float dot = 0.0f;
    for (int d = 0; d < DD; ++d) dot += sQ[d] * kr[d];
    sc = dot * INV_SCALE;
    if (mask[tid] > 0.5f) sc = NEG_INF;
  }
  sRed[tid] = sc;
  __syncthreads();
  for (int s = 128; s > 0; s >>= 1) {
    if (tid < s) sRed[tid] = fmaxf(sRed[tid], sRed[tid + s]);
    __syncthreads();
  }
  const float mx = sRed[0];
  __syncthreads();
  float ev = (tid < S) ? __expf(sc - mx) : 0.0f;
  sW[tid] = ev;
  sRed[tid] = ev;
  __syncthreads();
  for (int s = 128; s > 0; s >>= 1) {
    if (tid < s) sRed[tid] += sRed[tid + s];
    __syncthreads();
  }
  const float inv_sum = 1.0f / sRed[0];
  __syncthreads();
  sW[tid] = ev * inv_sum;
  __syncthreads();

  for (int d = tid; d < Dout; d += 256) {
    float a = 0.0f;
    for (int s = 0; s < S; ++s) a += sW[s] * Vv[(size_t)s * vstride + d];
    if (kind == 0) {
      rule_feed[b * 1024 + d] = a;
      inp_bf[b * 3072 + 512 + d] = (bfloat)a;
    } else if (kind == 1) {
      word_feed[b * 1024 + d] = a;
      inp_bf[b * 3072 + 2048 + d] = (bfloat)a;
    } else if (kind == 2) {
      r2w_feed[b * 512 + d] = a;
    } else {
      w2r_feed[b * 512 + d] = a;
    }
  }
}

// ---------------------------------------------------------------------------
// Readout: pres = tanh(inp @ roW.T).  M=16, N=512 per decoder, K=3072.
// grid 64: dec = bx>>5, column tile = bx&31. 1 wave per block.
// ---------------------------------------------------------------------------
__global__ __launch_bounds__(32) void readout_kernel(
    const bfloat* __restrict__ inp_bf,
    const bfloat* __restrict__ rro, const bfloat* __restrict__ wro,
    bfloat* __restrict__ rpres, bfloat* __restrict__ wpres, int t) {
  const int dec = blockIdx.x >> 5;
  const int jt = blockIdx.x & 31;
  const bfloat* W = dec ? wro : rro;
  bfloat* pres = dec ? wpres : rpres;
  const int lane = threadIdx.x;
  const int nbase = jt * 16;
  v8f acc = {};
  for (int k0 = 0; k0 < 3072; k0 += 32) {
    v16bf a = load_frag_a(inp_bf, 3072, k0, lane);
    v16bf b = load_frag_b(W, 3072, k0, lane, nbase);
    acc = wmma_bf16(a, b, acc);
  }
  const int sel = lane >> 4, nc = lane & 15;
#pragma unroll
  for (int r = 0; r < 8; ++r) {
    int b = r + 8 * sel;
    pres[((size_t)t * 16 + b) * DD + nbase + nc] = (bfloat)tanhf(acc[r]);
  }
}

// ---------------------------------------------------------------------------
// Shared A-tile staging for the logits kernels: 16x512 bf16 = 16 KB.
// Uses the async global->LDS path when available.
// ---------------------------------------------------------------------------
__device__ __forceinline__ void stage_pres_tile(const bfloat* __restrict__ pres,
                                                int t, bfloat* sA) {
#if HAS_ASYNC_LDS
  const char* g = (const char*)(pres + (size_t)t * 16 * 512);
  char* l = (char*)sA;
#pragma unroll
  for (int it = 0; it < 4; ++it) {       // 256 thr x 4 x 16B = 16 KB
    int off = (threadIdx.x + it * 256) * 16;
    async_copy_b128(g + off, l + off);
  }
  wait_async0();
#else
  const uint4* src = (const uint4*)(pres + (size_t)t * 16 * 512);
  uint4* dst = (uint4*)sA;
  for (int i = threadIdx.x; i < (16 * 512) / 8; i += 256) dst[i] = src[i];
#endif
  __syncthreads();
}

// ---------------------------------------------------------------------------
// Final logits GEMMs.  Word part: M-tile = t (16 batches), N = 32000, K = 512.
// ---------------------------------------------------------------------------
__global__ __launch_bounds__(256) void logits_word_kernel(
    const bfloat* __restrict__ pres,      // (T*16) x 512 bf16 (word)
    const bfloat* __restrict__ rout,      // VTOT x 512 bf16
    const int* __restrict__ score_mask,   // B x YL
    float* __restrict__ out) {
  __shared__ bfloat sA[16 * 512];
  const int t = blockIdx.x;
  stage_pres_tile(pres, t, sA);
  const int wv = threadIdx.x >> 5, lane = threadIdx.x & 31;
  const int nt = blockIdx.y * 8 + wv;     // < 2000
  const int nbase = nt * 16;
  v8f acc = {};
  for (int k0 = 0; k0 < 512; k0 += 32) {
    if (k0 + 32 < 512)
      __builtin_prefetch(rout + (size_t)(nbase + (lane & 15)) * 512 + k0 + 32, 0, 2);
    v16bf a = load_frag_a(sA, 512, k0, lane);
    v16bf b = load_frag_b(rout, 512, k0, lane, nbase);
    acc = wmma_bf16(a, b, acc);
  }
  const int sel = lane >> 4, nc = lane & 15;
  const int v = nbase + nc;
#pragma unroll
  for (int r = 0; r < 8; ++r) {
    int b = r + 8 * sel;
    int sm = score_mask[b * YLEN + t];
    float val = (sm == 0) ? NEG_INF : acc[r];   // word part masked when sm==0
    out[(size_t)b * YLEN * VTOT + (size_t)t * VTOT + v] = val;
  }
}

__global__ __launch_bounds__(256) void logits_rule_kernel(
    const bfloat* __restrict__ pres,      // (T*16) x 512 bf16 (rule)
    const bfloat* __restrict__ rout,      // VTOT x 512 bf16
    const int* __restrict__ score_mask,
    float* __restrict__ out) {
  __shared__ bfloat sA[16 * 512];
  const int t = blockIdx.x;
  stage_pres_tile(pres, t, sA);
  const int wv = threadIdx.x >> 5, lane = threadIdx.x & 31;
  const int nt = blockIdx.y * 8 + wv;
  if (nt >= 13) return;                    // ceil(200/16)=13 tiles (wave-uniform)
  const int nbase = nt * 16;
  v8f acc = {};
  for (int k0 = 0; k0 < 512; k0 += 32) {
    v16bf a = load_frag_a(sA, 512, k0, lane);
    int n = nbase + (lane & 15);
    if (n > RVN - 1) n = RVN - 1;          // clamp OOB loads of last tile
    int sel0 = (lane >> 4) & 1;
    v16bf b = *(const v16bf*)(rout + (size_t)(WVN + n) * 512 + k0 + sel0 * 16);
    acc = wmma_bf16(a, b, acc);
  }
  const int sel = lane >> 4, nc = lane & 15;
  const int col = nbase + nc;
#pragma unroll
  for (int r = 0; r < 8; ++r) {
    int b = r + 8 * sel;
    if (col < RVN) {
      int sm = score_mask[b * YLEN + t];
      float val = (sm == 1) ? NEG_INF : acc[r];  // rule part masked when sm==1
      out[(size_t)b * YLEN * VTOT + (size_t)t * VTOT + (WVN + col)] = val;
    }
  }
}

// ---------------------------------------------------------------------------
// Host orchestration
// ---------------------------------------------------------------------------
extern "C" void kernel_launch(void* const* d_in, const int* in_sizes, int n_in,
                              void* d_out, int out_size, void* d_ws, size_t ws_size,
                              hipStream_t stream) {
  (void)in_sizes; (void)n_in; (void)out_size; (void)ws_size;
  const float* x_enc      = (const float*)d_in[0];
  const float* x_enc_k    = (const float*)d_in[1];
  const float* dec_init_h = (const float*)d_in[2];
  const float* dec_init_c = (const float*)d_in[3];
  const float* x_mask     = (const float*)d_in[4];
  const int*   y_train    = (const int*)d_in[5];
  const int*   score_mask = (const int*)d_in[6];
  const float* emb_W      = (const float*)d_in[7];
  const float* rule_Wih   = (const float*)d_in[8];
  const float* rule_Whh   = (const float*)d_in[9];
  const float* rule_bih   = (const float*)d_in[10];
  const float* rule_bhh   = (const float*)d_in[11];
  const float* word_Wih   = (const float*)d_in[12];
  const float* word_Whh   = (const float*)d_in[13];
  const float* word_bih   = (const float*)d_in[14];
  const float* word_bhh   = (const float*)d_in[15];
  const float* rule_ro_W  = (const float*)d_in[16];
  const float* word_ro_W  = (const float*)d_in[17];
  const float* readout_W  = (const float*)d_in[18];
  float* out = (float*)d_out;

  // ---- bf16 workspace layout (element offsets) ----
  bfloat* bw = (bfloat*)d_ws;
  const size_t O_RWIH  = 0;                            // 2048*2048
  const size_t O_RWHH  = O_RWIH  + 2048ull * 2048;     // 2048*512
  const size_t O_WWIH  = O_RWHH  + 2048ull * 512;
  const size_t O_WWHH  = O_WWIH  + 2048ull * 2048;
  const size_t O_RRO   = O_WWHH  + 2048ull * 512;      // 512*3072
  const size_t O_WRO   = O_RRO   + 512ull * 3072;
  const size_t O_ROUT  = O_WRO   + 512ull * 3072;      // 32200*512
  const size_t O_WIN   = O_ROUT  + (size_t)VTOT * 512; // 16*2048
  const size_t O_RIN   = O_WIN   + 16ull * 2048;
  const size_t O_WHB   = O_RIN   + 16ull * 2048;       // 16*512
  const size_t O_RHB   = O_WHB   + 16ull * 512;
  const size_t O_INP   = O_RHB   + 16ull * 512;        // 16*3072
  const size_t O_RPRES = O_INP   + 16ull * 3072;       // 128*16*512
  const size_t O_WPRES = O_RPRES + 128ull * 16 * 512;
  const size_t BF_TOTAL = O_WPRES + 128ull * 16 * 512;

  // ---- f32 workspace after the bf16 region ----
  float* fw = (float*)((char*)d_ws + ((BF_TOTAL * 2 + 255) & ~(size_t)255));
  const size_t F_RH = 0, F_RC = 8192, F_WH = 16384, F_WC = 24576;
  const size_t F_RFEED = 32768, F_WFEED = 49152;
  const size_t F_R2W = 65536, F_W2R = 73728;
  const size_t F_RST = 81920, F_WST = F_RST + 1048576ull;
  const size_t F_RSM = F_WST + 1048576ull, F_WSM = F_RSM + 2048;
  const size_t ZERO_N = 16384 + 16384 + 8192 + 8192 + 2ull * 1048576; // feeds+states

  // ---- one-time (per call): convert weights to bf16, init carry state ----
  cvt_bf16_kernel<<<2048, 256, 0, stream>>>(bw + O_RWIH, rule_Wih, 2048ll * 2048);
  cvt_bf16_kernel<<<2048, 256, 0, stream>>>(bw + O_RWHH, rule_Whh, 2048ll * 512);
  cvt_bf16_kernel<<<2048, 256, 0, stream>>>(bw + O_WWIH, word_Wih, 2048ll * 2048);
  cvt_bf16_kernel<<<2048, 256, 0, stream>>>(bw + O_WWHH, word_Whh, 2048ll * 512);
  cvt_bf16_kernel<<<2048, 256, 0, stream>>>(bw + O_RRO,  rule_ro_W, 512ll * 3072);
  cvt_bf16_kernel<<<2048, 256, 0, stream>>>(bw + O_WRO,  word_ro_W, 512ll * 3072);
  cvt_bf16_kernel<<<2048, 256, 0, stream>>>(bw + O_ROUT, readout_W, (long)VTOT * 512);

  (void)hipMemcpyAsync(fw + F_RH, dec_init_h, 8192 * sizeof(float), hipMemcpyDeviceToDevice, stream);
  (void)hipMemcpyAsync(fw + F_RC, dec_init_c, 8192 * sizeof(float), hipMemcpyDeviceToDevice, stream);
  (void)hipMemcpyAsync(fw + F_WH, dec_init_h, 8192 * sizeof(float), hipMemcpyDeviceToDevice, stream);
  (void)hipMemcpyAsync(fw + F_WC, dec_init_c, 8192 * sizeof(float), hipMemcpyDeviceToDevice, stream);
  zero_f32_kernel<<<2048, 256, 0, stream>>>(fw + F_RFEED, (long)ZERO_N);
  smask_init_kernel<<<8, 256, 0, stream>>>(fw + F_RSM, fw + F_WSM);

  // ---- sequential decode ----
  for (int t = 0; t < YLEN; ++t) {
    prep_kernel<<<(81952 + 255) / 256, 256, 0, stream>>>(
        emb_W, y_train, fw + F_RFEED, fw + F_WFEED, fw + F_R2W, fw + F_W2R,
        fw + F_RH, fw + F_WH, bw + O_WIN, bw + O_RIN, bw + O_WHB, bw + O_RHB,
        fw + F_RSM, fw + F_WSM, t);
    // rule LSTM
    lstm_kernel<<<32, 128, 0, stream>>>(
        bw + O_RIN, bw + O_RHB, bw + O_RWIH, bw + O_RWHH, rule_bih, rule_bhh,
        fw + F_RH, fw + F_RC, fw + F_RST, bw + O_INP, /*inp_off=*/0,
        y_train, t, /*is_word=*/0);
    // word LSTM
    lstm_kernel<<<32, 128, 0, stream>>>(
        bw + O_WIN, bw + O_WHB, bw + O_WWIH, bw + O_WWHH, word_bih, word_bhh,
        fw + F_WH, fw + F_WC, fw + F_WST, bw + O_INP, /*inp_off=*/1536,
        y_train, t, /*is_word=*/1);
    attn_kernel<<<64, 256, 0, stream>>>(
        fw + F_RH, fw + F_WH, x_enc, x_enc_k, x_mask, fw + F_RST, fw + F_WST,
        fw + F_RSM, fw + F_WSM, fw + F_RFEED, fw + F_WFEED, fw + F_R2W,
        fw + F_W2R, bw + O_INP);
    readout_kernel<<<64, 32, 0, stream>>>(
        bw + O_INP, bw + O_RRO, bw + O_WRO, bw + O_RPRES, bw + O_WPRES, t);
  }

  // ---- final logits GEMMs ----
  logits_word_kernel<<<dim3(YLEN, 250), 256, 0, stream>>>(
      bw + O_WPRES, bw + O_ROUT, score_mask, out);
  logits_rule_kernel<<<dim3(YLEN, 2), 256, 0, stream>>>(
      bw + O_RPRES, bw + O_ROUT, score_mask, out);
}
